// GraphOfShots_11690900979944
// MI455X (gfx1250) — matrix operand order repaired and checked by reading, not compiled, not run on hardware
//
#include <hip/hip_runtime.h>

#define M_TOT 8192
#define NSUP  2048
#define NQRY  6144
#define EMB   256
#define KTOP  32
#define NWAY  10
#define EPS_BN 1e-5f
#define NEG_INF (-__builtin_inff())

typedef __attribute__((ext_vector_type(2)))  float  v2f;
typedef __attribute__((ext_vector_type(8)))  float  v8f;
typedef __attribute__((ext_vector_type(16))) __bf16 v16bf;

__device__ __forceinline__ unsigned short f2bf(float x) {
  unsigned u = __float_as_uint(x);
  unsigned r = u + 0x7FFFu + ((u >> 16) & 1u);   // round-to-nearest-even
  return (unsigned short)(r >> 16);
}

// ---------------- row normalize + bf16 convert ----------------
__global__ void k_rownorm(const float* __restrict__ z, unsigned short* __restrict__ zb) {
  int row = blockIdx.x, t = threadIdx.x;
  __shared__ float red[256];
  float v = z[(size_t)row * EMB + t];
  red[t] = v * v; __syncthreads();
  for (int s = 128; s > 0; s >>= 1) { if (t < s) red[t] += red[t + s]; __syncthreads(); }
  float rn = rsqrtf(red[0] + 1e-24f);
  zb[(size_t)row * EMB + t] = f2bf(v * rn);
}

// ---------------- fused affinity (bf16 WMMA, double-buffered B) + per-row top-K ----------------
// block: 256 thr = 8 waves; block owns 16 rows; wave w sweeps cols w*16, w*16+128, ...
__global__ void __launch_bounds__(256) k_aff_topk(const unsigned short* __restrict__ zb,
                                                  float* __restrict__ wv_out,
                                                  int* __restrict__ idx_out) {
  __shared__ float tkv[8][16][KTOP];
  __shared__ int   tki[8][16][KTOP];
  __shared__ float thr[8][16];
  int tid = threadIdx.x;
  int wid = tid >> 5, lane = tid & 31, hlf = lane >> 4, l = lane & 15;
  int rowbase = blockIdx.x * 16;

  for (int i = tid; i < 8 * 16 * KTOP; i += 256) { ((float*)tkv)[i] = NEG_INF; ((int*)tki)[i] = 0; }
  for (int i = tid; i < 128; i += 256) ((float*)thr)[i] = NEG_INF;
  __syncthreads();

  // A panel: 16 rows x K=256 held in registers as 8 chunks of K=32
  v16bf a[8];
  {
    const unsigned short* rp = zb + (size_t)(rowbase + l) * EMB;
    #pragma unroll
    for (int kk = 0; kk < 8; ++kk) {
      int base = kk * 32 + hlf * 8;   // lanes 0-15: K 0..7 / 16..23 ; lanes 16-31: K 8..15 / 24..31
      union { v16bf v; uint4 q[2]; } u;
      u.q[0] = *(const uint4*)(rp + base);
      u.q[1] = *(const uint4*)(rp + base + 16);
      a[kk] = u.v;
    }
  }

  union BU { v16bf v; uint4 q[2]; };
  BU b0[8], b1[8];

  auto loadB = [&](BU* b, int cbase) {
    const unsigned short* cp = zb + (size_t)(cbase + l) * EMB;
    #pragma unroll
    for (int kk = 0; kk < 8; ++kk) {
      int koff = kk * 32 + hlf * 8;   // B: lanes 0-15 K 0..7 / 16..23 ; lanes 16-31 K 8..15 / 24..31
      b[kk].q[0] = *(const uint4*)(cp + koff);
      b[kk].q[1] = *(const uint4*)(cp + koff + 16);
    }
  };
  auto wmma8 = [&](BU* b) {
    v8f acc = {};
    #pragma unroll
    for (int kk = 0; kk < 8; ++kk)
      acc = __builtin_amdgcn_wmma_f32_16x16x32_bf16(false, a[kk], false, b[kk].v,
                                                    (short)0, acc, false, false);
    return acc;
  };
  // top-K update: VGPR j -> row j (lanes 0-15) / row j+8 (lanes 16-31), col = cb+l
  auto processTile = [&](v8f acc, int cb) {
    int gcol = cb + l;
    #pragma unroll
    for (int j = 0; j < 8; ++j) {
      int rl = j + hlf * 8;
      int grow = rowbase + rl;
      float val = acc[j];
      if (gcol == grow) val = NEG_INF;               // mask diagonal
      float th = thr[wid][rl];
      unsigned bal = (unsigned)__ballot(val > th);
      bal = hlf ? (bal & 0xFFFF0000u) : (bal & 0x0000FFFFu);
      while (bal) {
        int src = __ffs(bal) - 1; bal &= bal - 1;
        float v = __shfl(val, src);
        int  ci = __shfl(gcol, src);
        float* TV = tkv[wid][rl];
        // 16-lane cooperative argmin over 32 entries
        float e0 = TV[l], e1 = TV[l + 16];
        float mn; int mp;
        if (e0 <= e1) { mn = e0; mp = l; } else { mn = e1; mp = l + 16; }
        #pragma unroll
        for (int off = 8; off; off >>= 1) {
          float ov = __shfl_xor(mn, off); int op = __shfl_xor(mp, off);
          if (ov < mn || (ov == mn && op < mp)) { mn = ov; mp = op; }
        }
        if (v > mn) {
          if (l == 0) { TV[mp] = v; tki[wid][rl][mp] = ci; }
          float nm = fminf(TV[l], TV[l + 16]);
          #pragma unroll
          for (int off = 8; off; off >>= 1) nm = fminf(nm, __shfl_xor(nm, off));
          if (l == 0) thr[wid][rl] = nm;
        }
      }
    }
  };

  // software-pipelined column sweep: 64 tiles/wave, ping-pong b0/b1
  loadB(b0, wid * 16);
  for (int cb = wid * 16; cb < M_TOT; cb += 256) {
    if (cb + 384 < M_TOT) {
      const unsigned short* np = zb + (size_t)(cb + 384 + l) * EMB;
      __builtin_prefetch(np, 0, 3);
      __builtin_prefetch(np + 128, 0, 3);
    }
    loadB(b1, cb + 128);                 // in flight while b0 WMMAs run
    v8f acc = wmma8(b0);
    processTile(acc, cb);
    if (cb + 256 < M_TOT) loadB(b0, cb + 256);  // in flight while b1 WMMAs run
    acc = wmma8(b1);
    processTile(acc, cb + 128);
  }
  __syncthreads();

  // merge 8 per-wave lists (256 candidates) -> top 32, one half-wave per row
  int hw = tid >> 4;          // row 0..15
  int ll = tid & 15;
  int grow = rowbase + hw;
  float* outw = wv_out + (size_t)grow * KTOP;
  int*   outi = idx_out + (size_t)grow * KTOP;
  for (int r = 0; r < KTOP; ++r) {
    float bv = NEG_INF; int be = 0;
    #pragma unroll
    for (int t2 = 0; t2 < 16; ++t2) {
      int e = ll * 16 + t2;
      float v = tkv[e >> 5][hw][e & 31];
      if (v > bv) { bv = v; be = e; }
    }
    #pragma unroll
    for (int off = 8; off; off >>= 1) {
      float ov = __shfl_xor(bv, off); int oe = __shfl_xor(be, off);
      if (ov > bv || (ov == bv && oe < be)) { bv = ov; be = oe; }
    }
    if (ll == 0) {
      outw[r] = fmaxf(bv, 0.0f) + 1e-6f;             // relu(vals)+1e-6
      outi[r] = tki[be >> 5][hw][be & 31];
      tkv[be >> 5][hw][be & 31] = NEG_INF;
    }
  }
}

// ---------------- degree / dinv ----------------
__global__ void k_deg(const float* __restrict__ wv, const int* __restrict__ idx,
                      float* __restrict__ deg) {
  int i = blockIdx.x * blockDim.x + threadIdx.x;
  if (i >= M_TOT) return;
  float s = 1.0f;
  for (int k = 0; k < KTOP; ++k) s += wv[i * KTOP + k];
  unsafeAtomicAdd(&deg[i], s);
  for (int k = 0; k < KTOP; ++k) unsafeAtomicAdd(&deg[idx[i * KTOP + k]], wv[i * KTOP + k]);
}
__global__ void k_dinv(const float* __restrict__ deg, float* __restrict__ dinv) {
  int i = blockIdx.x * blockDim.x + threadIdx.x;
  if (i < M_TOT) dinv[i] = rsqrtf(deg[i]);
}

// ---------------- xw = x @ W[0:256] (+ W[256+label] for support rows), fp32 WMMA ----------------
__global__ void __launch_bounds__(256) k_xw(const float* __restrict__ x, const float* __restrict__ W,
                                            const int* __restrict__ labels, int has_label,
                                            float* __restrict__ out) {
  int tid = threadIdx.x, wid = tid >> 5, lane = tid & 31, hlf = lane >> 4, l = lane & 15;
  int rowbase = blockIdx.x * 16;
  int ncol = blockIdx.y * 128 + wid * 16 + l;
  const float* xr = x + (size_t)(rowbase + l) * EMB;
  v8f acc = {};
  for (int k0 = 0; k0 < EMB; k0 += 16) {
    // batch 4 K-steps of loads, then 4 WMMAs (hide L2 latency behind matrix ops)
    v2f av[4], bv[4];
    #pragma unroll
    for (int u = 0; u < 4; ++u) {
      int kb = k0 + u * 4 + hlf * 2;    // A: lanes 0-15 K=kb,kb+1 ; 16-31 K=kb(+2),kb+3
      av[u] = *(const v2f*)(xr + kb);
      bv[u].x = W[(size_t)(kb + 0) * EMB + ncol];
      bv[u].y = W[(size_t)(kb + 1) * EMB + ncol];
    }
    #pragma unroll
    for (int u = 0; u < 4; ++u)
      acc = __builtin_amdgcn_wmma_f32_16x16x4_f32(false, av[u], false, bv[u],
                                                  (short)0, acc, false, false);
  }
  #pragma unroll
  for (int j = 0; j < 8; ++j) {
    int row = rowbase + j + hlf * 8;
    float v = acc[j];
    if (has_label && row < NSUP) v += W[(size_t)(EMB + labels[row]) * EMB + ncol];
    out[(size_t)row * EMB + ncol] = v;
  }
}

// ---------------- GCN conv: gather + native-f32-atomic scatter + self term ----------------
__global__ void k_conv(const float* __restrict__ xw, const float* __restrict__ wv,
                       const int* __restrict__ idx, const float* __restrict__ dinv,
                       float* __restrict__ out) {
  int i = blockIdx.x, f = threadIdx.x;
  float xwi = xw[(size_t)i * EMB + f];
  float di = dinv[i];
  float acc = 0.0f;
  for (int k = 0; k < KTOP; ++k) {
    int j = idx[i * KTOP + k];
    float c = di * wv[i * KTOP + k] * dinv[j];
    acc += c * xw[(size_t)j * EMB + f];                    // edge (j -> i)
    unsafeAtomicAdd(&out[(size_t)j * EMB + f], c * xwi);   // edge (i -> j)
  }
  unsafeAtomicAdd(&out[(size_t)i * EMB + f], acc + xwi * di * di);  // + xw/deg
}

// ---------------- batchnorm (2-stage column stats) ----------------
__global__ void k_bn_stats(const float* __restrict__ x, float* __restrict__ p1, float* __restrict__ p2) {
  int b = blockIdx.x, c = threadIdx.x;
  float s = 0.f, s2 = 0.f;
  for (int r = 0; r < 64; ++r) {
    float v = x[(size_t)(b * 64 + r) * EMB + c];
    s += v; s2 += v * v;
  }
  p1[b * EMB + c] = s; p2[b * EMB + c] = s2;
}
__global__ void k_bn_finalize(const float* __restrict__ p1, const float* __restrict__ p2,
                              const float* __restrict__ gamma, const float* __restrict__ beta,
                              float* __restrict__ scale, float* __restrict__ shift) {
  int c = threadIdx.x;
  float s = 0.f, s2 = 0.f;
  for (int b = 0; b < 128; ++b) { s += p1[b * EMB + c]; s2 += p2[b * EMB + c]; }
  float mu = s / (float)M_TOT;
  float var = fmaxf(s2 / (float)M_TOT - mu * mu, 0.0f);
  float sc = gamma[c] * rsqrtf(var + EPS_BN);
  scale[c] = sc; shift[c] = beta[c] - mu * sc;   // bias b cancels through BN
}
__global__ void k_bn_apply(float* __restrict__ x, const float* __restrict__ scale,
                           const float* __restrict__ shift, int relu) {
  int row = blockIdx.x, c = threadIdx.x;
  float v = x[(size_t)row * EMB + c] * scale[c] + shift[c];
  if (relu) v = fmaxf(v, 0.0f);
  x[(size_t)row * EMB + c] = v;
}

// ---------------- prototypes / logits / loss ----------------
__global__ void k_protos(const float* __restrict__ z, const int* __restrict__ slab,
                         float* __restrict__ protos, float* __restrict__ pnorm) {
  int c = blockIdx.x, f = threadIdx.x;
  float acc = 0.f; int cnt = 0;
  for (int i = 0; i < NSUP; ++i) {
    int li = slab[i];
    if (li == c) { acc += z[(size_t)i * EMB + f]; cnt++; }
  }
  float pr = acc / fmaxf((float)cnt, 1.0f);
  protos[c * EMB + f] = pr;
  __shared__ float red[256];
  red[f] = pr * pr; __syncthreads();
  for (int s = 128; s > 0; s >>= 1) { if (f < s) red[f] += red[f + s]; __syncthreads(); }
  if (f == 0) pnorm[c] = red[0];
}

__global__ void k_logits(const float* __restrict__ z, const float* __restrict__ protos,
                         const float* __restrict__ pnorm, const int* __restrict__ qlab,
                         float* __restrict__ out, float* __restrict__ nll) {
  int q = blockIdx.x, t = threadIdx.x;
  __shared__ float red[256];
  __shared__ float res[NWAY + 1];
  __shared__ float lg[NWAY];
  const float* zq = z + (size_t)(NSUP + q) * EMB;
  float zt = zq[t];
  red[t] = zt * zt; __syncthreads();
  for (int s = 128; s > 0; s >>= 1) { if (t < s) red[t] += red[t + s]; __syncthreads(); }
  if (t == 0) res[NWAY] = red[0];
  __syncthreads();
  for (int p = 0; p < NWAY; ++p) {
    red[t] = zt * protos[p * EMB + t]; __syncthreads();
    for (int s = 128; s > 0; s >>= 1) { if (t < s) red[t] += red[t + s]; __syncthreads(); }
    if (t == 0) res[p] = red[0];
    __syncthreads();
  }
  if (t < NWAY) {
    float d2 = res[NWAY] - 2.0f * res[t] + pnorm[t];
    float lgt = -sqrtf(fmaxf(d2, 1e-12f));
    lg[t] = lgt;
    out[(size_t)q * NWAY + t] = lgt;
  }
  __syncthreads();
  if (t == 0) {
    float m = NEG_INF;
    for (int p = 0; p < NWAY; ++p) m = fmaxf(m, lg[p]);
    float se = 0.0f;
    for (int p = 0; p < NWAY; ++p) se += __expf(lg[p] - m);
    float lse = m + __logf(se);
    nll[q] = lse - lg[qlab[q]];
  }
}

__global__ void k_loss(const float* __restrict__ nll, float* __restrict__ out) {
  int t = threadIdx.x;
  __shared__ float red[256];
  float s = 0.f;
  for (int i = t; i < NQRY; i += 256) s += nll[i];
  red[t] = s; __syncthreads();
  for (int k = 128; k > 0; k >>= 1) { if (t < k) red[t] += red[t + k]; __syncthreads(); }
  if (t == 0) out[0] = red[0] / (float)NQRY;
}

// ---------------- host orchestration ----------------
extern "C" void kernel_launch(void* const* d_in, const int* in_sizes, int n_in,
                              void* d_out, int out_size, void* d_ws, size_t ws_size,
                              hipStream_t stream) {
  (void)in_sizes; (void)n_in; (void)out_size; (void)ws_size;
  const float* h_s = (const float*)d_in[0];
  const float* h_q = (const float*)d_in[1];
  const int* slab = (const int*)d_in[2];
  const int* qlab = (const int*)d_in[3];
  // jax pytree flatten: per layer dict sorted keys -> W, b, beta, gamma (b unused: BN cancels it)
  const float *Wm[2][2], *Beta[2][2], *Gamma[2][2];
  int p = 4;
  for (int r = 0; r < 2; ++r)
    for (int li = 0; li < 2; ++li) {
      Wm[r][li]    = (const float*)d_in[p + 0];
      Beta[r][li]  = (const float*)d_in[p + 2];
      Gamma[r][li] = (const float*)d_in[p + 3];
      p += 4;
    }

  char* w = (char*)d_ws;
  size_t off = 0;
  auto alloc = [&](size_t bytes) { void* q = w + off; off = (off + bytes + 255) & ~(size_t)255; return q; };
  float* bufA = (float*)alloc((size_t)M_TOT * EMB * 4);
  float* bufB = (float*)alloc((size_t)M_TOT * EMB * 4);
  float* xw   = (float*)alloc((size_t)M_TOT * EMB * 4);
  unsigned short* zb = (unsigned short*)alloc((size_t)M_TOT * EMB * 2);
  float* wv   = (float*)alloc((size_t)M_TOT * KTOP * 4);
  int*   idx  = (int*)alloc((size_t)M_TOT * KTOP * 4);
  float* deg  = (float*)alloc(M_TOT * 4);
  float* dinv = (float*)alloc(M_TOT * 4);
  float* p1   = (float*)alloc(128 * EMB * 4);
  float* p2   = (float*)alloc(128 * EMB * 4);
  float* scale = (float*)alloc(EMB * 4);
  float* shift = (float*)alloc(EMB * 4);
  float* protos = (float*)alloc(NWAY * EMB * 4);
  float* pnorm  = (float*)alloc(256);
  float* nll    = (float*)alloc(NQRY * 4);   // total ~32 MB of ws

  hipMemcpyAsync(bufA, h_s, (size_t)NSUP * EMB * 4, hipMemcpyDeviceToDevice, stream);
  hipMemcpyAsync(bufA + (size_t)NSUP * EMB, h_q, (size_t)NQRY * EMB * 4, hipMemcpyDeviceToDevice, stream);

  for (int r = 0; r < 2; ++r) {
    k_rownorm<<<M_TOT, 256, 0, stream>>>(bufA, zb);
    k_aff_topk<<<M_TOT / 16, 256, 0, stream>>>(zb, wv, idx);
    hipMemsetAsync(deg, 0, M_TOT * 4, stream);
    k_deg<<<M_TOT / 256, 256, 0, stream>>>(wv, idx, deg);
    k_dinv<<<M_TOT / 256, 256, 0, stream>>>(deg, dinv);
    // layer 0 (with label-block term) -> bufB, BN+relu
    k_xw<<<dim3(M_TOT / 16, 2), 256, 0, stream>>>(bufA, Wm[r][0], slab, 1, xw);
    hipMemsetAsync(bufB, 0, (size_t)M_TOT * EMB * 4, stream);
    k_conv<<<M_TOT, 256, 0, stream>>>(xw, wv, idx, dinv, bufB);
    k_bn_stats<<<128, 256, 0, stream>>>(bufB, p1, p2);
    k_bn_finalize<<<1, 256, 0, stream>>>(p1, p2, Gamma[r][0], Beta[r][0], scale, shift);
    k_bn_apply<<<M_TOT, 256, 0, stream>>>(bufB, scale, shift, 1);
    // layer 1 -> bufA, BN (no relu)
    k_xw<<<dim3(M_TOT / 16, 2), 256, 0, stream>>>(bufB, Wm[r][1], slab, 0, xw);
    hipMemsetAsync(bufA, 0, (size_t)M_TOT * EMB * 4, stream);
    k_conv<<<M_TOT, 256, 0, stream>>>(xw, wv, idx, dinv, bufA);
    k_bn_stats<<<128, 256, 0, stream>>>(bufA, p1, p2);
    k_bn_finalize<<<1, 256, 0, stream>>>(p1, p2, Gamma[r][1], Beta[r][1], scale, shift);
    k_bn_apply<<<M_TOT, 256, 0, stream>>>(bufA, scale, shift, 0);
  }

  float* out = (float*)d_out;
  k_protos<<<NWAY, 256, 0, stream>>>(bufA, slab, protos, pnorm);
  k_logits<<<NQRY, 256, 0, stream>>>(bufA, protos, pnorm, qlab, out, nll);
  k_loss<<<1, 256, 0, stream>>>(nll, out + (size_t)NQRY * NWAY);
}
